// CrossAttentionLayer_28673201668275
// MI455X (gfx1250) — matrix operand order repaired and checked by reading, not compiled, not run on hardware
//
#include <hip/hip_runtime.h>
#include <hip/hip_bf16.h>

typedef _Float16 half_t;
typedef __attribute__((ext_vector_type(16))) _Float16 v16h;
typedef __attribute__((ext_vector_type(8)))  float    v8f;

#define WMMA_F32_F16(a, b, c) \
  __builtin_amdgcn_wmma_f32_16x16x32_f16(false, (a), false, (b), (short)0, (c), false, false)

// Problem constants (from the reference instance).
constexpr int D   = 256;   // d_model
constexpr int H   = 8;     // heads
constexpr int HD  = 32;    // head dim
constexpr int NB  = 16;    // batch
constexpr int L   = 4096;  // keys per sample
constexpr int NQ  = 100;   // queries per sample
constexpr int NQP = 112;   // padded to 7 tiles of 16

// ---------------------------------------------------------------------------
// WMMA operand loaders (wave32; contiguous k order used by this file for both A and B).
// A 16x32 (MxK): lane m=lane&15; halves j=0..15 map to K = 16*hi + j.
//                K pairs per VGPR.
// B 32x16 (KxN): lane n=lane&15; halves j=0..15 map to K = 16*hi + j.
// C/D 16x16 f32: VGPR g -> M = g + 8*hi, N = lane&15.
// ---------------------------------------------------------------------------
__device__ inline v16h load_A_rm(const half_t* __restrict__ base, int stride, int lane) {
  const int m = lane & 15, hi = lane >> 4;
  const half_t* row = base + m * stride + 16 * hi;
  v16h a;
#pragma unroll
  for (int v = 0; v < 8; ++v) {
    const int kb = 2 * v;
    a[2 * v]     = row[kb];
    a[2 * v + 1] = row[kb + 1];
  }
  return a;
}

__device__ inline v16h load_B_rm(const half_t* __restrict__ base, int stride, int lane) {
  const int n = lane & 15, hi = lane >> 4;
  const half_t* row = base + n * stride + 16 * hi;
  v16h b;
#pragma unroll
  for (int j = 0; j < 16; ++j) b[j] = row[j];
  return b;
}

// ---------------------------------------------------------------------------
// Kernel 0: fp32 -> f16 weight conversion.
// ---------------------------------------------------------------------------
__global__ void f32_to_f16_kernel(const float* __restrict__ src, half_t* __restrict__ dst, int n) {
  const int i = blockIdx.x * blockDim.x + threadIdx.x;
  if (i < n) dst[i] = (half_t)src[i];
}

// ---------------------------------------------------------------------------
// Kernel 1: fused K/V projection over the ragged source gather.
// Grid: NB * (L/32) blocks; block = 256 threads (8 waves, wave == head).
// 32 tokens per block (two M-tiles) so every weight B-tile feeds 2 WMMAs.
// K stored (b,h,key,dim) f16; V stored transposed (b,h,dim,key) f16.
// ---------------------------------------------------------------------------
__global__ __launch_bounds__(256) void proj_kv_kernel(
    const float* __restrict__ src, const int* __restrict__ offs,
    const float* __restrict__ biasK, const float* __restrict__ biasV,
    const half_t* __restrict__ WkH, const half_t* __restrict__ WvH,
    half_t* __restrict__ Kh, half_t* __restrict__ Vt) {
  __shared__ half_t xs[32][D + 8];
  const int b    = blockIdx.x >> 7;          // L/32 == 128 tiles per sample
  const int p0   = (blockIdx.x & 127) << 5;
  const int base = offs[b];
  const int tid  = threadIdx.x;
  for (int i = tid; i < 32 * (D / 4); i += 256) {   // 2048 float4
    const int m = i >> 6, c4 = i & 63;
    const float4 f = reinterpret_cast<const float4*>(src + (size_t)(base + p0 + m) * D)[c4];
    xs[m][c4 * 4 + 0] = (half_t)f.x;
    xs[m][c4 * 4 + 1] = (half_t)f.y;
    xs[m][c4 * 4 + 2] = (half_t)f.z;
    xs[m][c4 * 4 + 3] = (half_t)f.w;
  }
  __syncthreads();

  const int wave = tid >> 5, lane = tid & 31;
  const int h = wave, n0 = h * HD;
  v8f cK[2][2] = {}, cV[2][2] = {};   // [m-tile][n-tile]
#pragma unroll
  for (int kk = 0; kk < D; kk += 32) {
    const v16h a0  = load_A_rm(&xs[0][0] + kk,  D + 8, lane);
    const v16h a1  = load_A_rm(&xs[16][0] + kk, D + 8, lane);
    const v16h bK0 = load_B_rm(WkH + (size_t)n0 * D + kk, D, lane);
    const v16h bK1 = load_B_rm(WkH + (size_t)(n0 + 16) * D + kk, D, lane);
    const v16h bV0 = load_B_rm(WvH + (size_t)n0 * D + kk, D, lane);
    const v16h bV1 = load_B_rm(WvH + (size_t)(n0 + 16) * D + kk, D, lane);
    cK[0][0] = WMMA_F32_F16(a0, bK0, cK[0][0]);
    cK[1][0] = WMMA_F32_F16(a1, bK0, cK[1][0]);
    cK[0][1] = WMMA_F32_F16(a0, bK1, cK[0][1]);
    cK[1][1] = WMMA_F32_F16(a1, bK1, cK[1][1]);
    cV[0][0] = WMMA_F32_F16(a0, bV0, cV[0][0]);
    cV[1][0] = WMMA_F32_F16(a1, bV0, cV[1][0]);
    cV[0][1] = WMMA_F32_F16(a0, bV1, cV[0][1]);
    cV[1][1] = WMMA_F32_F16(a1, bV1, cV[1][1]);
  }
  const int hi = lane >> 4, n = lane & 15;
  const float bk0f = biasK[n0 + n], bk1f = biasK[n0 + 16 + n];
  const float bv0f = biasV[n0 + n], bv1f = biasV[n0 + 16 + n];
  const size_t bh = (size_t)b * H + h;
#pragma unroll
  for (int mt = 0; mt < 2; ++mt) {
#pragma unroll
    for (int g = 0; g < 8; ++g) {
      const int t = p0 + 16 * mt + g + 8 * hi;
      Kh[(bh * L + t) * HD + n]      = (half_t)(cK[mt][0][g] + bk0f);
      Kh[(bh * L + t) * HD + 16 + n] = (half_t)(cK[mt][1][g] + bk1f);
      Vt[(bh * HD + n) * L + t]      = (half_t)(cV[mt][0][g] + bv0f);
      Vt[(bh * HD + 16 + n) * L + t] = (half_t)(cV[mt][1][g] + bv1f);
    }
  }
}

// ---------------------------------------------------------------------------
// Kernel 2: Q projection with 1/sqrt(HD) folded in; queries padded to 112.
// Grid: NB*7 blocks; wave == head. Qh layout (b,h,q,dim) f16.
// ---------------------------------------------------------------------------
__global__ __launch_bounds__(256) void proj_q_kernel(
    const float* __restrict__ query, const float* __restrict__ biasQ,
    const half_t* __restrict__ WqH, half_t* __restrict__ Qh) {
  __shared__ half_t xs[16][D + 8];
  const int b = blockIdx.x / 7, q0 = (blockIdx.x % 7) * 16;
  const int tid = threadIdx.x;
  for (int i = tid; i < 16 * (D / 4); i += 256) {
    const int m = i >> 6, c4 = i & 63;
    const int q = q0 + m;
    float4 f = make_float4(0.f, 0.f, 0.f, 0.f);
    if (q < NQ) f = reinterpret_cast<const float4*>(query + ((size_t)b * NQ + q) * D)[c4];
    xs[m][c4 * 4 + 0] = (half_t)f.x;
    xs[m][c4 * 4 + 1] = (half_t)f.y;
    xs[m][c4 * 4 + 2] = (half_t)f.z;
    xs[m][c4 * 4 + 3] = (half_t)f.w;
  }
  __syncthreads();

  const int wave = tid >> 5, lane = tid & 31;
  const int h = wave, n0 = h * HD;
  v8f c0 = {}, c1 = {};
#pragma unroll
  for (int kk = 0; kk < D; kk += 32) {
    const v16h a  = load_A_rm(&xs[0][0] + kk, D + 8, lane);
    const v16h b0 = load_B_rm(WqH + (size_t)n0 * D + kk, D, lane);
    const v16h b1 = load_B_rm(WqH + (size_t)(n0 + 16) * D + kk, D, lane);
    c0 = WMMA_F32_F16(a, b0, c0);
    c1 = WMMA_F32_F16(a, b1, c1);
  }
  const int hi = lane >> 4, n = lane & 15;
  const float scale = 0.17677669529663687f;  // 1/sqrt(32)
  const float b0f = biasQ[n0 + n], b1f = biasQ[n0 + 16 + n];
  const size_t bh = (size_t)b * H + h;
#pragma unroll
  for (int g = 0; g < 8; ++g) {
    const int q = q0 + g + 8 * hi;
    Qh[(bh * NQP + q) * HD + n]      = (half_t)((c0[g] + b0f) * scale);
    Qh[(bh * NQP + q) * HD + 16 + n] = (half_t)((c1[g] + b1f) * scale);
  }
}

// ---------------------------------------------------------------------------
// Kernel 3: flash attention per (b, h, q-tile). 8 waves split the 4096 keys,
// each wave runs online softmax over its range; LDS merge at the end.
// Row-sums come from a WMMA against an all-ones B matrix (acc2), so the
// normalizer sums exactly the f16-rounded P used for P·V.
// ctx written fp32, layout (b, q_padded, D).
// ---------------------------------------------------------------------------
__global__ __launch_bounds__(256) void attn_kernel(
    const half_t* __restrict__ Qh, const half_t* __restrict__ Kh,
    const half_t* __restrict__ Vt, float* __restrict__ ctx) {
  const int qt = blockIdx.x % 7;
  const int bh = blockIdx.x / 7;  // 0..127
  const int tid = threadIdx.x, wave = tid >> 5, lane = tid & 31;
  const int hi = lane >> 4, ln = lane & 15;

  __shared__ half_t pS[8][16][32];     // per-wave P staging (A-layout source)
  __shared__ float  lm[8][16], ll[8][16];
  __shared__ float  lctx[8][16][32];

  const v16h aQ = load_A_rm(Qh + ((size_t)bh * NQP + qt * 16) * HD, HD, lane);

  v16h bOnes;
#pragma unroll
  for (int j = 0; j < 16; ++j) bOnes[j] = (half_t)1.0f;

  float mrow[8];
#pragma unroll
  for (int g = 0; g < 8; ++g) mrow[g] = -1e30f;
  v8f acc0 = {}, acc1 = {}, accS = {};   // ctx dims 0..15, 16..31, row-sum

  const half_t* Kbase = Kh + (size_t)bh * L * HD;
  const half_t* Vbase = Vt + (size_t)bh * HD * L;

  const int kEnd = (wave + 1) * (L / 8);
  for (int k0 = wave * (L / 8); k0 < kEnd; k0 += 32) {
    // scores for 32 keys: S = Q (16x32) x K^T (32x16) twice
    const v16h bK0 = load_B_rm(Kbase + (size_t)k0 * HD, HD, lane);
    const v16h bK1 = load_B_rm(Kbase + (size_t)(k0 + 16) * HD, HD, lane);
    v8f s0 = {}, s1 = {};
    s0 = WMMA_F32_F16(aQ, bK0, s0);
    s1 = WMMA_F32_F16(aQ, bK1, s1);

#pragma unroll
    for (int g = 0; g < 8; ++g) {
      float x = fmaxf(s0[g], s1[g]);
#pragma unroll
      for (int d = 1; d < 16; d <<= 1) x = fmaxf(x, __shfl_xor(x, d, 32));
      const float mnew = fmaxf(mrow[g], x);
      const float sc = __expf(mrow[g] - mnew);
      const float p0 = __expf(s0[g] - mnew);
      const float p1 = __expf(s1[g] - mnew);
      mrow[g] = mnew;
      acc0[g] *= sc;
      acc1[g] *= sc;
      accS[g] *= sc;
      // stage P(q=g+8*hi, key) into per-wave LDS (row-major 16x32 f16)
      pS[wave][g + 8 * hi][ln]      = (half_t)p0;
      pS[wave][g + 8 * hi][16 + ln] = (half_t)p1;
    }
    // intra-wave LDS store->load ordering (no cross-wave sharing of pS[wave])
    asm volatile("s_wait_dscnt 0" ::: "memory");
    __builtin_amdgcn_wave_barrier();

    const v16h aP  = load_A_rm(&pS[wave][0][0], 32, lane);
    const v16h bV0 = load_B_rm(Vbase + k0, L, lane);                    // dims 0..15
    const v16h bV1 = load_B_rm(Vbase + (size_t)16 * L + k0, L, lane);   // dims 16..31
    acc0 = WMMA_F32_F16(aP, bV0, acc0);
    acc1 = WMMA_F32_F16(aP, bV1, acc1);
    accS = WMMA_F32_F16(aP, bOnes, accS);   // row-sums of P (all columns equal)
  }

  // publish per-wave partials
  if (ln == 0) {
#pragma unroll
    for (int g = 0; g < 8; ++g) {
      lm[wave][g + 8 * hi] = mrow[g];
      ll[wave][g + 8 * hi] = accS[g];
    }
  }
#pragma unroll
  for (int g = 0; g < 8; ++g) {
    lctx[wave][g + 8 * hi][ln]      = acc0[g];
    lctx[wave][g + 8 * hi][16 + ln] = acc1[g];
  }
  __syncthreads();

  const int b = bh >> 3, h = bh & 7;
  for (int i = tid; i < 16 * 32; i += 256) {
    const int q = i >> 5, d = i & 31;
    float M = -1e30f;
#pragma unroll
    for (int w = 0; w < 8; ++w) M = fmaxf(M, lm[w][q]);
    float Ltot = 0.f, X = 0.f;
#pragma unroll
    for (int w = 0; w < 8; ++w) {
      const float e = __expf(lm[w][q] - M);
      Ltot += e * ll[w][q];
      X    += e * lctx[w][q][d];
    }
    const int qq = qt * 16 + q;
    ctx[((size_t)b * NQP + qq) * D + h * HD + d] = X / Ltot;
  }
}

// ---------------------------------------------------------------------------
// Kernel 4: out = ctx @ Wo^T + bo + query (residual), fp32 output.
// ---------------------------------------------------------------------------
__global__ __launch_bounds__(256) void out_proj_kernel(
    const float* __restrict__ ctxbuf, const float* __restrict__ biasO,
    const float* __restrict__ query, const half_t* __restrict__ WoH,
    float* __restrict__ out) {
  __shared__ half_t xs[16][D + 8];
  const int b = blockIdx.x / 7, q0 = (blockIdx.x % 7) * 16;
  const int tid = threadIdx.x;
  for (int i = tid; i < 16 * (D / 4); i += 256) {
    const int m = i >> 6, c4 = i & 63;
    const float4 f =
        reinterpret_cast<const float4*>(ctxbuf + ((size_t)b * NQP + q0 + m) * D)[c4];
    xs[m][c4 * 4 + 0] = (half_t)f.x;
    xs[m][c4 * 4 + 1] = (half_t)f.y;
    xs[m][c4 * 4 + 2] = (half_t)f.z;
    xs[m][c4 * 4 + 3] = (half_t)f.w;
  }
  __syncthreads();

  const int wave = tid >> 5, lane = tid & 31;
  const int n0 = wave * 32;
  v8f c0 = {}, c1 = {};
#pragma unroll
  for (int kk = 0; kk < D; kk += 32) {
    const v16h a  = load_A_rm(&xs[0][0] + kk, D + 8, lane);
    const v16h b0 = load_B_rm(WoH + (size_t)n0 * D + kk, D, lane);
    const v16h b1 = load_B_rm(WoH + (size_t)(n0 + 16) * D + kk, D, lane);
    c0 = WMMA_F32_F16(a, b0, c0);
    c1 = WMMA_F32_F16(a, b1, c1);
  }
  const int hi = lane >> 4, n = lane & 15;
  const float b0f = biasO[n0 + n], b1f = biasO[n0 + 16 + n];
#pragma unroll
  for (int g = 0; g < 8; ++g) {
    const int q = q0 + g + 8 * hi;
    if (q < NQ) {
      const size_t r = ((size_t)b * NQ + q) * D;
      out[r + n0 + n]      = c0[g] + b0f + query[r + n0 + n];
      out[r + n0 + 16 + n] = c1[g] + b1f + query[r + n0 + 16 + n];
    }
  }
}

// ---------------------------------------------------------------------------
// Launch. Inputs: source, query, batch_offsets, Wq, bq, Wk, bk, Wv, bv, Wo, bo.
// ---------------------------------------------------------------------------
extern "C" void kernel_launch(void* const* d_in, const int* in_sizes, int n_in,
                              void* d_out, int out_size, void* d_ws, size_t ws_size,
                              hipStream_t stream) {
  (void)in_sizes; (void)n_in; (void)out_size; (void)ws_size;
  const float* source = (const float*)d_in[0];
  const float* query  = (const float*)d_in[1];
  const int*   offs   = (const int*)d_in[2];
  const float* Wq = (const float*)d_in[3];
  const float* bq = (const float*)d_in[4];
  const float* Wk = (const float*)d_in[5];
  const float* bk = (const float*)d_in[6];
  const float* Wv = (const float*)d_in[7];
  const float* bv = (const float*)d_in[8];
  const float* Wo = (const float*)d_in[9];
  const float* bo = (const float*)d_in[10];
  float* out = (float*)d_out;

  // Workspace layout (bytes).
  char* ws = (char*)d_ws;
  half_t* WqH = (half_t*)(ws + 0);                 // 128 KB each
  half_t* WkH = (half_t*)(ws + (1u << 17));
  half_t* WvH = (half_t*)(ws + 2u * (1u << 17));
  half_t* WoH = (half_t*)(ws + 3u * (1u << 17));
  half_t* Qh  = (half_t*)(ws + 524288u);           // NB*H*NQP*HD halves = 896 KB
  float*  ctxb = (float*)(ws + 1441792u);          // NB*NQP*D fp32 = 1.75 MB
  half_t* Kh  = (half_t*)(ws + 3276800u);          // NB*H*L*HD halves = 32 MB
  half_t* Vt  = (half_t*)(ws + 36831232u);         // 32 MB

  const int nW = D * D;
  const int cb = (nW + 255) / 256;
  f32_to_f16_kernel<<<cb, 256, 0, stream>>>(Wq, WqH, nW);
  f32_to_f16_kernel<<<cb, 256, 0, stream>>>(Wk, WkH, nW);
  f32_to_f16_kernel<<<cb, 256, 0, stream>>>(Wv, WvH, nW);
  f32_to_f16_kernel<<<cb, 256, 0, stream>>>(Wo, WoH, nW);

  proj_kv_kernel<<<NB * (L / 32), 256, 0, stream>>>(source, offs, bk, bv, WkH, WvH, Kh, Vt);
  proj_q_kernel<<<NB * 7, 256, 0, stream>>>(query, bq, WqH, Qh);
  attn_kernel<<<NB * H * 7, 256, 0, stream>>>(Qh, Kh, Vt, ctxb);
  out_proj_kernel<<<NB * 7, 256, 0, stream>>>(ctxb, bo, query, WoH, out);
}
